// BiLstmCrf_60155311948414
// MI455X (gfx1250) — matrix operand order repaired and checked by reading, not compiled, not run on hardware
//
#include <hip/hip_runtime.h>

#define B_ 64
#define L_ 2048
#define T_ 128

typedef __attribute__((ext_vector_type(16))) _Float16 v16h;
typedef __attribute__((ext_vector_type(8)))  _Float16 v8h;
typedef __attribute__((ext_vector_type(8)))  float    v8f;

// Raw-hardware transcendentals: v_exp_f32 / v_log_f32 are base-2.
__device__ __forceinline__ float fast_exp(float x) {
  return __builtin_amdgcn_exp2f(x * 1.4426950408889634f);
}
__device__ __forceinline__ float fast_log(float x) {
  return __builtin_amdgcn_logf(x) * 0.6931471805599453f;
}

// ---------------------------------------------------------------------------
// Kernel 1: gold-path (numerator) score per batch -> num[b]
// ---------------------------------------------------------------------------
__global__ __launch_bounds__(256) void crf_numerator(
    const float* __restrict__ em, const int* __restrict__ tgt,
    const int* __restrict__ mask, const float* __restrict__ start,
    const float* __restrict__ endt, const float* __restrict__ trans,
    float* __restrict__ num)
{
  __shared__ float ssum[256];
  __shared__ int   smask[256];
  const int b   = blockIdx.x;
  const int tid = threadIdx.x;
  const int*   tb = tgt  + (size_t)b * L_;
  const int*   mb = mask + (size_t)b * L_;
  const float* eb = em   + (size_t)b * L_ * T_;

  float s = 0.f;
  int msum = 0;
  for (int t = tid; t < L_; t += 256) msum += (mb[t] != 0);
  for (int t = 1 + tid; t < L_; t += 256) {
    int tp = tb[t - 1], tc = tb[t];
    float c = trans[tp * T_ + tc] + eb[(size_t)t * T_ + tc];
    s += (mb[t] != 0) ? c : 0.f;
  }
  ssum[tid] = s; smask[tid] = msum;
  __syncthreads();
  for (int off = 128; off > 0; off >>= 1) {
    if (tid < off) { ssum[tid] += ssum[tid + off]; smask[tid] += smask[tid + off]; }
    __syncthreads();
  }
  if (tid == 0) {
    float r = ssum[0];
    int t0 = tb[0];
    r += start[t0] + eb[t0];              // em[b,0,t0]
    int send = smask[0] - 1;              // seq_ends
    r += endt[tb[send]];
    num[b] = r;
  }
}

// ---------------------------------------------------------------------------
// Kernel 2: forward algorithm (denominator) via exp-domain WMMA scan
//   grid = 4 blocks x 256 threads (8 waves); block handles 16 batch rows.
//   Wave w owns N-tile (output tags) [16w, 16w+16).
// ---------------------------------------------------------------------------
__global__ __launch_bounds__(256) void crf_forward(
    const float* __restrict__ em, const int* __restrict__ mask,
    const float* __restrict__ start, const float* __restrict__ endt,
    const float* __restrict__ trans, float* __restrict__ den)
{
  __shared__ __align__(16) _Float16 P[16 * T_];   // exp(score-rowmax) f16
  __shared__ __align__(16) float    sc[16 * T_];  // running forward scores
  __shared__ __align__(16) float    rowmax[16];
  __shared__ __align__(16) int      smrow[16];    // mask[b, t] per row

  const int tid  = threadIdx.x;
  const int wave = tid >> 5;                 // 0..7  -> N tile
  const int lane = tid & 31;
  const int lrow = lane & 15;
  const int hi   = lane >> 4;                // 0/1 half-wave
  const int b0   = blockIdx.x * 16;
  const int ncol = wave * 16 + lrow;         // output tag column of this lane

  // ---- static B fragments: Bfrag[c][i] = exp(trans[k][ncol]),
  //      k = 32c + 8*hi + i (i<8), 32c + 16 + 8*hi + (i-8) (i>=8)
  v16h Bfrag[4];
  #pragma unroll
  for (int c = 0; c < 4; ++c) {
    #pragma unroll
    for (int i = 0; i < 16; ++i) {
      int k = c * 32 + (i < 8 ? hi * 8 + i : 16 + hi * 8 + (i - 8));
      Bfrag[c][i] = (_Float16)fast_exp(trans[k * T_ + ncol]);
    }
  }

  // ---- init: score0 = start + em[:,0,:]
  for (int i = tid; i < 16 * T_; i += 256) {
    int r = i >> 7, j = i & (T_ - 1);
    sc[i] = start[j] + em[((size_t)(b0 + r) * L_) * T_ + j];
  }
  __syncthreads();

  // per-lane private copy of the score cells this lane owns (C layout:
  // VGPR v -> row v + 8*hi, col ncol) -- avoids re-reading old value.
  float myscore[8];
  #pragma unroll
  for (int v = 0; v < 8; ++v) myscore[v] = sc[(v + 8 * hi) * T_ + ncol];

  const int myrow = tid >> 4;   // 0..15 (16 lanes cooperate per row)
  const int seg   = tid & 15;

  for (int t = 1; t < L_; ++t) {
    // ---- fused phase: rowmax + P = exp(score - rowmax)
    float4 a0 = *(const float4*)&sc[myrow * T_ + seg * 8];
    float4 a1 = *(const float4*)&sc[myrow * T_ + seg * 8 + 4];
    float m = fmaxf(fmaxf(fmaxf(a0.x, a0.y), fmaxf(a0.z, a0.w)),
                    fmaxf(fmaxf(a1.x, a1.y), fmaxf(a1.z, a1.w)));
    #pragma unroll
    for (int off = 1; off < 16; off <<= 1) m = fmaxf(m, __shfl_xor(m, off, 32));
    if (seg == 0) {
      rowmax[myrow] = m;
      smrow[myrow]  = mask[(size_t)(b0 + myrow) * L_ + t];
    }
    v8h pv;
    pv[0] = (_Float16)fast_exp(a0.x - m);
    pv[1] = (_Float16)fast_exp(a0.y - m);
    pv[2] = (_Float16)fast_exp(a0.z - m);
    pv[3] = (_Float16)fast_exp(a0.w - m);
    pv[4] = (_Float16)fast_exp(a1.x - m);
    pv[5] = (_Float16)fast_exp(a1.y - m);
    pv[6] = (_Float16)fast_exp(a1.z - m);
    pv[7] = (_Float16)fast_exp(a1.w - m);
    *(v8h*)&P[myrow * T_ + seg * 8] = pv;      // ds_store_b128
    __syncthreads();

    // ---- WMMA: acc[16x16] = P[16x128] x expT-tile[128x16] (K chained 4x32)
    v8f acc = {};
    #pragma unroll
    for (int c = 0; c < 4; ++c) {
      const _Float16* ap = &P[lrow * T_ + c * 32 + hi * 8];
      v8h alo = *(const v8h*)(ap);             // ds_load_b128
      v8h ahi = *(const v8h*)(ap + 16);        // ds_load_b128
      v16h afrag;
      #pragma unroll
      for (int i = 0; i < 8; ++i) { afrag[i] = alo[i]; afrag[i + 8] = ahi[i]; }
      acc = __builtin_amdgcn_wmma_f32_16x16x32_f16(
          false, afrag, false, Bfrag[c], (short)0, acc, false, false);
    }

    // ---- combine: score' = rowmax + log(acc) + em[b,t,n]; mask-select.
    float4 rm0 = *(const float4*)&rowmax[hi * 8];
    float4 rm1 = *(const float4*)&rowmax[hi * 8 + 4];
    int4   mk0 = *(const int4*)&smrow[hi * 8];
    int4   mk1 = *(const int4*)&smrow[hi * 8 + 4];
    float rmv[8] = {rm0.x, rm0.y, rm0.z, rm0.w, rm1.x, rm1.y, rm1.z, rm1.w};
    int   mkv[8] = {mk0.x, mk0.y, mk0.z, mk0.w, mk1.x, mk1.y, mk1.z, mk1.w};
    #pragma unroll
    for (int v = 0; v < 8; ++v) {
      int r = v + 8 * hi;
      size_t eoff = ((size_t)(b0 + r) * L_ + t) * T_ + ncol;
      float e    = em[eoff];
      float newv = rmv[v] + fast_log(acc[v]) + e;
      myscore[v] = (mkv[v] != 0) ? newv : myscore[v];
      sc[r * T_ + ncol] = myscore[v];          // ds_store_b32
      if (t + 1 < L_) __builtin_prefetch(&em[eoff + T_], 0, 0);
    }
    __syncthreads();
  }

  // ---- den[b] = logsumexp_j(score[b,j] + end[j])
  float vals[8];
  float m = -3.0e38f;
  #pragma unroll
  for (int j = 0; j < 8; ++j) {
    vals[j] = sc[myrow * T_ + seg * 8 + j] + endt[seg * 8 + j];
    m = fmaxf(m, vals[j]);
  }
  #pragma unroll
  for (int off = 1; off < 16; off <<= 1) m = fmaxf(m, __shfl_xor(m, off, 32));
  float s = 0.f;
  #pragma unroll
  for (int j = 0; j < 8; ++j) s += fast_exp(vals[j] - m);
  #pragma unroll
  for (int off = 1; off < 16; off <<= 1) s += __shfl_xor(s, off, 32);
  if (seg == 0) den[b0 + myrow] = m + fast_log(s);
}

// ---------------------------------------------------------------------------
// Kernel 3: out = sum_b (den[b] - num[b])   (deterministic tree reduce)
// ---------------------------------------------------------------------------
__global__ __launch_bounds__(64) void crf_reduce(
    const float* __restrict__ den, const float* __restrict__ num,
    float* __restrict__ out)
{
  __shared__ float s[64];
  int tid = threadIdx.x;
  s[tid] = den[tid] - num[tid];
  __syncthreads();
  for (int off = 32; off > 0; off >>= 1) {
    if (tid < off) s[tid] += s[tid + off];
    __syncthreads();
  }
  if (tid == 0) out[0] = s[0];
}

// ---------------------------------------------------------------------------
extern "C" void kernel_launch(void* const* d_in, const int* in_sizes, int n_in,
                              void* d_out, int out_size, void* d_ws, size_t ws_size,
                              hipStream_t stream) {
  const float* em    = (const float*)d_in[0];
  const int*   tgt   = (const int*)d_in[1];
  const int*   mask  = (const int*)d_in[2];
  const float* start = (const float*)d_in[3];
  const float* endt  = (const float*)d_in[4];
  const float* trans = (const float*)d_in[5];

  float* den = (float*)d_ws;      // 64 floats
  float* num = den + B_;          // 64 floats

  crf_numerator<<<B_, 256, 0, stream>>>(em, tgt, mask, start, endt, trans, num);
  crf_forward<<<B_ / 16, 256, 0, stream>>>(em, mask, start, endt, trans, den);
  crf_reduce<<<1, B_, 0, stream>>>(den, num, (float*)d_out);
}
